// GNNDenoiser_46849503264903
// MI455X (gfx1250) — compile-verified, matrix operand order checked by reading, not run on hardware
//
#include <hip/hip_runtime.h>
#include <hip/hip_bf16.h>
#include <stdint.h>

// ---------------------------------------------------------------------------
// GNN denoiser for MI455X (gfx1250, wave32, WMMA + async global->LDS copies).
// All GEMMs: bf16 A/B, fp32 accumulate via v_wmma_f32_16x16x32_bf16.
// K-loop: 3-stage LDS pipeline fed by global_load_async_to_lds_b128 (ASYNCcnt),
// one barrier per K-step; transfers overlap the WMMAs.
// Edge MLP layer-1 factored through nodes:  e_in@eW1 = P[row] + Q[col] + d*w_last.
// Edges processed in 8192-edge chunks; mean-aggregation exploits col[e] == e/8.
// ---------------------------------------------------------------------------

typedef unsigned short u16;
typedef __attribute__((ext_vector_type(16))) __bf16 v16bf;
typedef __attribute__((ext_vector_type(8)))  float  v8f;

#define H_DIM    2048
#define D_DIM    1024
#define N_NODES  4096
#define E_EDGES  32768
#define NBLK     4
#define CH_NODES 1024
#define CH_EDGES 8192

__device__ __forceinline__ u16 f2bf(float f) {
  uint32_t u = __float_as_uint(f);
  u += 0x7FFFu + ((u >> 16) & 1u);        // round-to-nearest-even
  return (u16)(u >> 16);
}
__device__ __forceinline__ float bf2f(u16 h) {
  return __uint_as_float(((uint32_t)h) << 16);
}
__device__ __forceinline__ float silu_f(float x) {
  return x / (1.0f + __expf(-x));
}

// LDS byte offset of a __shared__ object: on AMDGCN the generic (flat) address
// of LDS keeps the byte offset in addr[31:0], which is exactly what the async
// instruction's LDS-address VGPR operand consumes.
__device__ __forceinline__ unsigned lds_off(const void* p) {
  return (unsigned)(uintptr_t)p;
}

// CDNA5 async copy: 16B global -> LDS per lane, tracked with ASYNCcnt.
__device__ __forceinline__ void async_ld128(unsigned dst_lds, const void* src) {
  asm volatile("global_load_async_to_lds_b128 %0, %1, off"
               :: "v"(dst_lds), "v"(src)
               : "memory");
}

// ---------------------------------------------------------------------------
// WMMA GEMM: C[M,N] = act( A[M,K] @ Wt[N,K]^T + bias + add0 + add1 )
// A, Wt bf16 (row-major, Wt pre-transposed so K is contiguous for both).
// Block = 256 threads (8 waves), tile 128x128, K-step 32, 3-stage async LDS
// pipeline. Wave (wm 0..3, wn 0..1) owns a 32x64 sub-tile = 2x4 WMMA frags.
// ---------------------------------------------------------------------------
union FragU { uint4 u[2]; v16bf v; };

#define LROW 40            // 32 K-elems + 8 pad (keeps 16B alignment, no conflicts)
#define LTILE (128 * LROW) // u16 elements per (stage, matrix) tile

__global__ __launch_bounds__(256)
void k_gemm_wmma(const u16* __restrict__ A, const u16* __restrict__ Wt,
                 const float* __restrict__ bias,
                 const float* __restrict__ add0, const float* __restrict__ add1,
                 float* __restrict__ outF, u16* __restrict__ outB,
                 int M, int N, int K, int act)
{
  __shared__ __attribute__((aligned(16))) u16 smem[3][2][LTILE]; // [stage][A/B]

  const int tid   = threadIdx.x;
  const int wave  = tid >> 5;
  const int lane  = tid & 31;
  const int wm    = wave >> 1;      // 0..3 -> 32-row strip
  const int wn    = wave & 1;       // 0..1 -> 64-col strip
  const int half  = lane >> 4;      // K-half selector (WMMA lane layout)
  const int l16   = lane & 15;      // M (A) / N (B,C) index within fragment
  const int tileM = blockIdx.y * 128;
  const int tileN = blockIdx.x * 128;

  v8f acc[2][4] = {};

  const int ldRow = tid >> 2;       // 0..63
  const int ldK   = (tid & 3) * 8;  // 0,8,16,24

  // Issue one stage: 4 async b128 instructions per wave (A lo/hi, B lo/hi).
  auto issue = [&](int k0, int s) {
    const u16* gA = A  + (size_t)(tileM + ldRow) * K + k0 + ldK;
    const u16* gB = Wt + (size_t)(tileN + ldRow) * K + k0 + ldK;
    const unsigned dA = lds_off(&smem[s][0][ldRow * LROW + ldK]);
    const unsigned dB = lds_off(&smem[s][1][ldRow * LROW + ldK]);
    async_ld128(dA,                  gA);
    async_ld128(dA + 64 * LROW * 2, gA + (size_t)64 * K);
    async_ld128(dB,                  gB);
    async_ld128(dB + 64 * LROW * 2, gB + (size_t)64 * K);
  };

  // Consume one stage: 12 ds_load_b128 -> 8 WMMAs; drain DScnt so the next
  // barrier may safely release this buffer for async overwrite.
  auto compute = [&](int s) {
    const u16* sA = smem[s][0];
    const u16* sB = smem[s][1];
    FragU af[2], bq[4];
#pragma unroll
    for (int mi = 0; mi < 2; ++mi) {
      const int r = wm * 32 + mi * 16 + l16;
      af[mi].u[0] = *(const uint4*)(sA + r * LROW + half * 8);      // K half*8..
      af[mi].u[1] = *(const uint4*)(sA + r * LROW + 16 + half * 8); // K 16+half*8..
    }
#pragma unroll
    for (int ni = 0; ni < 4; ++ni) {
      const int c = wn * 64 + ni * 16 + l16;
      bq[ni].u[0] = *(const uint4*)(sB + c * LROW + half * 8);
      bq[ni].u[1] = *(const uint4*)(sB + c * LROW + 16 + half * 8);
    }
#pragma unroll
    for (int mi = 0; mi < 2; ++mi)
#pragma unroll
      for (int ni = 0; ni < 4; ++ni)
        acc[mi][ni] = __builtin_amdgcn_wmma_f32_16x16x32_bf16(
            false, af[mi].v, false, bq[ni].v,
            (short)0, acc[mi][ni], false, false);
    asm volatile("s_wait_dscnt 0x0" ::: "memory");
  };

  const int nk = K >> 5;             // K is a multiple of 32, nk >= 2
  issue(0, 0);
  issue(32, 1);

  int sIssue = 2, sComp = 0;
  for (int kk = 0; kk + 2 < nk; ++kk) {
    asm volatile("s_wait_asynccnt 0x4" ::: "memory");  // stage kk landed
    __syncthreads();                                   // visible to all waves;
                                                       // stage kk-1 reads done
    issue((kk + 2) << 5, sIssue);                      // overlaps the WMMAs
    compute(sComp);
    sIssue = (sIssue == 2) ? 0 : sIssue + 1;
    sComp  = (sComp  == 2) ? 0 : sComp  + 1;
  }
  // Tail: stages nk-2 and nk-1.
  asm volatile("s_wait_asynccnt 0x4" ::: "memory");
  __syncthreads();
  compute(sComp);
  sComp = (sComp == 2) ? 0 : sComp + 1;
  asm volatile("s_wait_asynccnt 0x0" ::: "memory");
  __syncthreads();
  compute(sComp);

  // Epilogue. C layout: lanes0-15 row=r, lanes16-31 row=8+r, col=l16.
#pragma unroll
  for (int mi = 0; mi < 2; ++mi) {
#pragma unroll
    for (int ni = 0; ni < 4; ++ni) {
      const int gn = tileN + wn * 64 + ni * 16 + l16;
      const float bv = bias ? bias[gn] : 0.0f;
#pragma unroll
      for (int r = 0; r < 8; ++r) {
        const int gm = tileM + wm * 32 + mi * 16 + half * 8 + r;
        const size_t idx = (size_t)gm * N + gn;
        float v = acc[mi][ni][r] + bv;
        if (add0) v += add0[idx];
        if (add1) v += add1[idx];
        if (act)  v = silu_f(v);
        if (outF) outF[idx] = v;
        if (outB) outB[idx] = f2bf(v);
      }
    }
  }
}

// ---------------------------------------------------------------------------
// LayerNorm over H=2048, optional fp32 + bf16 outputs.
// ---------------------------------------------------------------------------
__global__ __launch_bounds__(256)
void k_layernorm(const float* __restrict__ x, const float* __restrict__ g,
                 const float* __restrict__ b,
                 float* __restrict__ outF, u16* __restrict__ outB)
{
  __shared__ float rs[256];
  __shared__ float rq[256];
  const int row = blockIdx.x;
  const int tid = threadIdx.x;
  const float* xr = x + (size_t)row * H_DIM;
  float s = 0.f, q = 0.f;
  for (int i = tid; i < H_DIM; i += 256) { float v = xr[i]; s += v; q += v * v; }
  rs[tid] = s; rq[tid] = q;
  __syncthreads();
  for (int off = 128; off > 0; off >>= 1) {
    if (tid < off) { rs[tid] += rs[tid + off]; rq[tid] += rq[tid + off]; }
    __syncthreads();
  }
  const float mu  = rs[0] * (1.0f / H_DIM);
  const float var = rq[0] * (1.0f / H_DIM) - mu * mu;
  const float inv = rsqrtf(var + 1e-5f);
  for (int i = tid; i < H_DIM; i += 256) {
    const float v = (xr[i] - mu) * inv * g[i] + b[i];
    const size_t idx = (size_t)row * H_DIM + i;
    if (outF) outF[idx] = v;
    if (outB) outB[idx] = f2bf(v);
  }
}

// fp32 W[Kd,Nd] -> bf16 Wt[Nd,Kd] (transpose so GEMM reads contiguous K)
__global__ __launch_bounds__(256)
void k_convT(const float* __restrict__ W, u16* __restrict__ Wt, int Kd, int Nd)
{
  __shared__ float t[32][33];
  const int bk = blockIdx.x * 32;
  const int bn = blockIdx.y * 32;
  const int tx = threadIdx.x;   // 0..31
  const int ty = threadIdx.y;   // 0..7
  for (int j = 0; j < 32; j += 8)
    t[ty + j][tx] = W[(size_t)(bk + ty + j) * Nd + bn + tx];
  __syncthreads();
  for (int j = 0; j < 32; j += 8)
    Wt[(size_t)(bn + ty + j) * Kd + bk + tx] = f2bf(t[tx][ty + j]);
}

__global__ void k_f2bf(const float* __restrict__ in, u16* __restrict__ out, int n)
{
  const int i = blockIdx.x * blockDim.x + threadIdx.x;
  if (i < n) out[i] = f2bf(in[i]);
}

__global__ void k_dist(const float* __restrict__ gc, const int* __restrict__ row,
                       const int* __restrict__ col, float* __restrict__ dist, int E)
{
  const int e = blockIdx.x * blockDim.x + threadIdx.x;
  if (e >= E) return;
  const int r = row[e], c = col[e];
  const float dx = gc[r * 3 + 0] - gc[c * 3 + 0];
  const float dy = gc[r * 3 + 1] - gc[c * 3 + 1];
  const float dz = gc[r * 3 + 2] - gc[c * 3 + 2];
  dist[e] = sqrtf(dx * dx + dy * dy + dz * dz);
}

// m1[e_local,:] = bf16( silu( P[row[e]] + Q[col[e]] + dist[e]*w_last ) )
// (eb1 was folded into P's GEMM bias.)
__global__ __launch_bounds__(256)
void k_edge_pre(const float* __restrict__ P, const float* __restrict__ Q,
                const float* __restrict__ dist,
                const int* __restrict__ row, const int* __restrict__ col,
                const float* __restrict__ wlast, u16* __restrict__ m1, int e0)
{
  const int el = blockIdx.x;
  const int e  = e0 + el;
  const int r  = row[e], c = col[e];
  const float d = dist[e];
  const float* Pr = P + (size_t)r * H_DIM;
  const float* Qc = Q + (size_t)c * H_DIM;
  u16* mo = m1 + (size_t)el * H_DIM;
  for (int h = threadIdx.x; h < H_DIM; h += 256)
    mo[h] = f2bf(silu_f(Pr[h] + Qc[h] + d * wlast[h]));
}

// agg[n0+nl,:] = bf16( (1/8) * sum_{j<8} m2[nl*8+j,:] )   (deg == K == 8 always)
__global__ __launch_bounds__(256)
void k_agg(const u16* __restrict__ m2, u16* __restrict__ agg, int n0)
{
  const int nl = blockIdx.x;
  const u16* src = m2 + (size_t)nl * 8 * H_DIM;
  u16* dst = agg + (size_t)(n0 + nl) * H_DIM;
  for (int h = threadIdx.x; h < H_DIM; h += 256) {
    float s = 0.f;
#pragma unroll
    for (int j = 0; j < 8; ++j) s += bf2f(src[(size_t)j * H_DIM + h]);
    dst[h] = f2bf(s * 0.125f);
  }
}

// ---------------------------------------------------------------------------
extern "C" void kernel_launch(void* const* d_in, const int* in_sizes, int n_in,
                              void* d_out, int out_size, void* d_ws, size_t ws_size,
                              hipStream_t stream)
{
  const float* y    = (const float*)d_in[0];
  const float* gc   = (const float*)d_in[1];
  const int*   ei   = (const int*)  d_in[2];
  const float* inW  = (const float*)d_in[3];
  const float* inb  = (const float*)d_in[4];
  const float* eW1  = (const float*)d_in[5];
  const float* eb1  = (const float*)d_in[6];
  const float* eW2  = (const float*)d_in[7];
  const float* eb2  = (const float*)d_in[8];
  const float* nW1  = (const float*)d_in[9];
  const float* nb1  = (const float*)d_in[10];
  const float* nW2  = (const float*)d_in[11];
  const float* nb2  = (const float*)d_in[12];
  const float* mW1  = (const float*)d_in[13];
  const float* mb1  = (const float*)d_in[14];
  const float* mW2  = (const float*)d_in[15];
  const float* mb2  = (const float*)d_in[16];
  const float* ln1g = (const float*)d_in[17];
  const float* ln1b = (const float*)d_in[18];
  const float* ln2g = (const float*)d_in[19];
  const float* ln2b = (const float*)d_in[20];
  const float* outW = (const float*)d_in[21];
  const float* outb = (const float*)d_in[22];
  float* out = (float*)d_out;

  const int* rowI = ei;
  const int* colI = ei + E_EDGES;

  // ---- workspace carve-up (256B aligned) ----
  char* ws = (char*)d_ws;
  size_t off = 0;
  auto carve = [&](size_t bytes) -> char* {
    char* p = ws + off;
    off += (bytes + 255) & ~(size_t)255;
    return p;
  };
  const size_t NH4 = (size_t)N_NODES * H_DIM * 4;
  const size_t NH2 = (size_t)N_NODES * H_DIM * 2;

  float* h    = (float*)carve(NH4);
  float* hn   = (float*)carve(NH4);
  float* P    = (float*)carve(NH4);            // later reused as tmpF
  float* Q    = (float*)carve(NH4);
  u16*   hnb  = (u16*)  carve(NH2);
  u16*   aggb = (u16*)  carve(NH2);
  u16*   b1   = (u16*)  carve(NH2);
  u16*   b2   = (u16*)  carve(NH2);
  u16*   ybf  = (u16*)  carve((size_t)N_NODES * D_DIM * 2);
  u16*   wt   = (u16*)  carve((size_t)H_DIM * H_DIM * 2);
  u16*   m1   = (u16*)  carve((size_t)CH_EDGES * H_DIM * 2);
  u16*   m2   = (u16*)  carve((size_t)CH_EDGES * H_DIM * 2);
  float* dist = (float*)carve((size_t)E_EDGES * 4);

  auto gemm = [&](const u16* A, const u16* W_, const float* bias,
                  const float* a0, const float* a1,
                  float* oF, u16* oB, int M, int N, int K, int act) {
    dim3 g(N / 128, M / 128);
    k_gemm_wmma<<<g, 256, 0, stream>>>(A, W_, bias, a0, a1, oF, oB, M, N, K, act);
  };
  auto convT = [&](const float* W_, int Kd, int Nd) {
    k_convT<<<dim3(Kd / 32, Nd / 32), dim3(32, 8), 0, stream>>>(W_, wt, Kd, Nd);
  };

  // ---- pipeline ----
  k_f2bf<<<(N_NODES * D_DIM + 255) / 256, 256, 0, stream>>>(y, ybf, N_NODES * D_DIM);
  k_dist<<<(E_EDGES + 255) / 256, 256, 0, stream>>>(gc, rowI, colI, dist, E_EDGES);

  // h = y @ in_W + in_b
  convT(inW, D_DIM, H_DIM);
  gemm(ybf, wt, inb, nullptr, nullptr, h, nullptr, N_NODES, H_DIM, D_DIM, 0);

  const size_t HH = (size_t)H_DIM * H_DIM;
  for (int i = 0; i < NBLK; ++i) {
    const float* eW1i = eW1 + (size_t)i * (2 * H_DIM + 1) * H_DIM;
    const float* eW2i = eW2 + (size_t)i * HH;
    const float* nW1i = nW1 + (size_t)i * 2 * HH;
    const float* nW2i = nW2 + (size_t)i * HH;
    const float* mW1i = mW1 + (size_t)i * HH;
    const float* mW2i = mW2 + (size_t)i * HH;

    // hn = LN(h) (fp32 + bf16)
    k_layernorm<<<N_NODES, 256, 0, stream>>>(h, ln1g + i * H_DIM, ln1b + i * H_DIM, hn, hnb);

    // Factored edge layer 1: P = hn@eW1[:H] + eb1 ; Q = hn@eW1[H:2H]
    convT(eW1i, H_DIM, H_DIM);
    gemm(hnb, wt, eb1 + i * H_DIM, nullptr, nullptr, P, nullptr, N_NODES, H_DIM, H_DIM, 0);
    convT(eW1i + HH, H_DIM, H_DIM);
    gemm(hnb, wt, nullptr, nullptr, nullptr, Q, nullptr, N_NODES, H_DIM, H_DIM, 0);
    convT(eW2i, H_DIM, H_DIM);

    // Edge chunks: gather+silu -> GEMM(silu) -> mean-aggregate (deg==8)
    for (int c = 0; c < E_EDGES / CH_EDGES; ++c) {
      const int e0 = c * CH_EDGES, n0 = c * CH_NODES;
      k_edge_pre<<<CH_EDGES, 256, 0, stream>>>(P, Q, dist, rowI, colI,
                                               eW1i + 2 * HH, m1, e0);
      gemm(m1, wt, eb2 + i * H_DIM, nullptr, nullptr, nullptr, m2,
           CH_EDGES, H_DIM, H_DIM, 1);
      k_agg<<<CH_NODES, 256, 0, stream>>>(m2, aggb, n0);
    }

    // d1 = silu(hn@nW1a + agg@nW1b + nb1)   (tmpF aliases P, now free)
    float* tmpF = P;
    convT(nW1i, H_DIM, H_DIM);
    gemm(hnb, wt, nb1 + i * H_DIM, nullptr, nullptr, tmpF, nullptr, N_NODES, H_DIM, H_DIM, 0);
    convT(nW1i + HH, H_DIM, H_DIM);
    gemm(aggb, wt, nullptr, tmpF, nullptr, nullptr, b1, N_NODES, H_DIM, H_DIM, 1);

    // h = h0 + hn + (d1 @ nW2 + nb2)
    convT(nW2i, H_DIM, H_DIM);
    gemm(b1, wt, nb2 + i * H_DIM, h, hn, h, nullptr, N_NODES, H_DIM, H_DIM, 0);

    // h = h + silu(LN2(h) @ mW1 + mb1) @ mW2 + mb2
    k_layernorm<<<N_NODES, 256, 0, stream>>>(h, ln2g + i * H_DIM, ln2b + i * H_DIM,
                                             nullptr, b1);
    convT(mW1i, H_DIM, H_DIM);
    gemm(b1, wt, mb1 + i * H_DIM, nullptr, nullptr, nullptr, b2, N_NODES, H_DIM, H_DIM, 1);
    convT(mW2i, H_DIM, H_DIM);
    gemm(b2, wt, mb2 + i * H_DIM, h, nullptr, h, nullptr, N_NODES, H_DIM, H_DIM, 0);
  }

  // out = h @ out_W + out_b
  k_f2bf<<<(N_NODES * H_DIM + 255) / 256, 256, 0, stream>>>(h, b1, N_NODES * H_DIM);
  convT(outW, H_DIM, D_DIM);
  gemm(b1, wt, outb, nullptr, nullptr, out, nullptr, N_NODES, D_DIM, H_DIM, 0);
}
